// PatchWiseTemporalAttention_43585328119982
// MI455X (gfx1250) — matrix operand order repaired and checked by reading, not compiled, not run on hardware
//
#include <hip/hip_runtime.h>
#include <hip/hip_bf16.h>

// ---------------------------------------------------------------------------
// PatchWiseTemporalAttention for MI455X (gfx1250, wave32, WMMA bf16 + TDM)
//   B=4, T=32, L=256, D=1024, H=4, Dh=64, inner=256
//   1. convert_w : w_qkv / w_out -> bf16 (kept in native N x K layout)
//   2. ln        : LayerNorm over D + transpose to (B*L*T, D) + bf16 cast
//   3. qkv_gemm  : TDM-stage 64x1024 B-panel to LDS (128KB), WMMA bf16, +bias
//   4. rope      : RoPE on q,k; q/k/v -> bf16 head layout [(bl*H+h), t, d]
//   5. attn      : per (bl,h): QK^T wmma -> softmax(+mask) -> PV wmma -> bf16
//   6. out_gemm  : TDM-stage 64x256 B-panel to LDS (32KB), WMMA bf16, scatter
// ---------------------------------------------------------------------------

typedef __bf16 bf16_t;
typedef __attribute__((address_space(3))) __bf16 lds_bf16_t;
typedef __attribute__((ext_vector_type(16))) __bf16 v16bf;
typedef __attribute__((ext_vector_type(8)))  float  v8f;
typedef __attribute__((ext_vector_type(4)))  unsigned int v4u;
typedef __attribute__((ext_vector_type(8)))  int v8i;
typedef __attribute__((ext_vector_type(4)))  int v4i;

#define B_      4
#define T_      32
#define L_      256
#define D_      1024
#define H_      4
#define DH_     64
#define INNER_  256
#define BL_     (B_ * L_)      // 1024
#define ROWS_   (BL_ * T_)     // 32768
#define NQKV_   768

// ---------------- WMMA fragment helpers (CDNA5 16x16x32 bf16 layouts) ------

__device__ __forceinline__ v16bf load_A_frag(const bf16_t* __restrict__ A,
                                             int lda, int m0, int k0, int lane) {
  const int m  = m0 + (lane & 15);
  const int kb = (lane >= 16) ? 16 : 0;
  v16bf f;
#pragma unroll
  for (int j = 0; j < 8; ++j) {
    const int koff = 2 * j;
    const int k = k0 + kb + koff;
    f[2 * j]     = A[(size_t)m * lda + k];
    f[2 * j + 1] = A[(size_t)m * lda + k + 1];
  }
  return f;
}

// B[k][n] = src[n*ld + k] (source (N x K) row-major). Per-lane data is 16
// contiguous bf16 -> two 16-byte loads.
__device__ __forceinline__ v16bf load_Bt_frag(const bf16_t* __restrict__ src,
                                              int ld, int k0, int n0, int lane) {
  const int n  = n0 + (lane & 15);
  const int kb = (lane >= 16) ? 16 : 0;
  v16bf f;
#pragma unroll
  for (int j = 0; j < 8; ++j) {
    const int k = k0 + kb + 2 * j;
    f[2 * j]     = src[(size_t)n * ld + k];
    f[2 * j + 1] = src[(size_t)n * ld + k + 1];
  }
  return f;
}

// Same as load_Bt_frag but reading an LDS (addrspace(3)) panel -> ds_load.
__device__ __forceinline__ v16bf load_Bt_frag_lds(const lds_bf16_t* src,
                                                  int ld, int k0, int n0,
                                                  int lane) {
  const int n  = n0 + (lane & 15);
  const int kb = (lane >= 16) ? 16 : 0;
  v16bf f;
#pragma unroll
  for (int j = 0; j < 8; ++j) {
    const int k = k0 + kb + 2 * j;
    f[2 * j]     = src[n * ld + k];
    f[2 * j + 1] = src[n * ld + k + 1];
  }
  return f;
}

// B-fragment from (K x N) row-major storage (used for V in attention).
__device__ __forceinline__ v16bf load_B_frag(const bf16_t* __restrict__ Bt,
                                             int ldb, int k0, int n0, int lane) {
  const int n  = n0 + (lane & 15);
  const int kb = (lane >= 16) ? 16 : 0;
  v16bf f;
#pragma unroll
  for (int j = 0; j < 8; ++j) {
    const int k = k0 + kb + 2 * j;
    f[2 * j]     = Bt[(size_t)k * ldb + n];
    f[2 * j + 1] = Bt[(size_t)(k + 1) * ldb + n];
  }
  return f;
}

__device__ __forceinline__ v8f wmma_bf16(v16bf a, v16bf b, v8f c) {
  return __builtin_amdgcn_wmma_f32_16x16x32_bf16(
      false, a, false, b, (short)0, c, false, false);
}

// Opaque LDS pointer at byte offset `off` (laundered through inline asm so the
// compiler cannot fold TDM-written LDS reads to undef; masking restores
// 128-byte alignment knowledge for b128 ds loads).
__device__ __forceinline__ const lds_bf16_t* lds_panel_ptr(unsigned off) {
  asm volatile("" : "+v"(off) :: "memory");
  off &= ~127u;
  return (const lds_bf16_t*)off;
}

// ---------------- Tensor Data Mover: 2D bf16 panel -> LDS -------------------
// D# per CDNA5 ISA ch.8: count=1, type=2, data_size=2B, 2D tile, no options.
// 6-arg builtin: (uint32x4 g0, int32x8 g1, int32x4 g2, int32x4 g3,
//                 int32x8 gx, i32 cpol)

__device__ __forceinline__ void tdm_load_panel(const bf16_t* gsrc,
                                               unsigned lds_byte_off,
                                               unsigned tensor_w, unsigned tensor_h,
                                               unsigned tile_w, unsigned tile_h,
                                               unsigned row_stride_elems) {
  const unsigned long long ga = (unsigned long long)(uintptr_t)gsrc;
  v4u g0;
  g0[0] = 1u;                                          // count=1
  g0[1] = lds_byte_off;                                // lds_addr
  g0[2] = (unsigned)(ga & 0xffffffffull);              // global_addr[31:0]
  g0[3] = (unsigned)((ga >> 32) & 0x1ffffffull) | (2u << 30);  // addr hi | type=2
  v8i g1;
  g1[0] = (int)(1u << 16);                             // data_size=1 (2 bytes)
  g1[1] = (int)((tensor_w & 0xffffu) << 16);           // tensor_dim0 lo16
  g1[2] = (int)(((tensor_w >> 16) & 0xffffu) | ((tensor_h & 0xffffu) << 16));
  g1[3] = (int)(((tensor_h >> 16) & 0xffffu) | ((tile_w & 0xffffu) << 16));
  g1[4] = (int)(tile_h & 0xffffu);                     // tile_dim1 (tile_dim2=0)
  g1[5] = (int)row_stride_elems;                       // tensor_dim0_stride lo32
  g1[6] = 0;
  g1[7] = 0;
  const v4i g2 = {0, 0, 0, 0};
  const v4i g3 = {0, 0, 0, 0};
  const v8i gx = {0, 0, 0, 0, 0, 0, 0, 0};
  __builtin_amdgcn_tensor_load_to_lds(g0, g1, g2, g3, gx, 0);
}

// ---------------- 1. weight convert (bf16 cast, native layout) --------------

__global__ void __launch_bounds__(256)
convert_w_kernel(const float* __restrict__ w_qkv, const float* __restrict__ w_out,
                 bf16_t* __restrict__ wqkvB, bf16_t* __restrict__ woutB) {
  const int id = blockIdx.x * blockDim.x + threadIdx.x;
  if (id < NQKV_ * D_)  wqkvB[id] = (bf16_t)w_qkv[id];
  if (id < D_ * INNER_) woutB[id] = (bf16_t)w_out[id];
}

// ---------------- 2. LayerNorm + transpose + bf16 cast ----------------------

__global__ void __launch_bounds__(256)
ln_kernel(const float* __restrict__ x, const float* __restrict__ gamma,
          const float* __restrict__ beta, bf16_t* __restrict__ xn) {
  __shared__ float ssum[256];
  __shared__ float ssq[256];
  const int row = blockIdx.x;            // row = ((b*T + t)*L + l)
  const int l = row & (L_ - 1);
  const int t = (row >> 8) & (T_ - 1);
  const int b = row >> 13;
  const int tid = threadIdx.x;
  const float* xr = x + (size_t)row * D_;

  float v[4], s = 0.f, q = 0.f;
#pragma unroll
  for (int i = 0; i < 4; ++i) {
    v[i] = xr[tid + i * 256];
    s += v[i];
    q += v[i] * v[i];
  }
  ssum[tid] = s; ssq[tid] = q;
  __syncthreads();
  for (int off = 128; off > 0; off >>= 1) {
    if (tid < off) { ssum[tid] += ssum[tid + off]; ssq[tid] += ssq[tid + off]; }
    __syncthreads();
  }
  const float mean = ssum[0] * (1.0f / D_);
  const float var  = ssq[0] * (1.0f / D_) - mean * mean;
  const float inv  = rsqrtf(var + 1e-5f);

  const size_t orow = ((size_t)(b * L_ + l) * T_ + t) * D_;
#pragma unroll
  for (int i = 0; i < 4; ++i) {
    const int d = tid + i * 256;
    xn[orow + d] = (bf16_t)((v[i] - mean) * inv * gamma[d] + beta[d]);
  }
}

// ---------------- 3. QKV GEMM (TDM-staged B panel) --------------------------

__global__ void __launch_bounds__(256)
qkv_gemm_kernel(const bf16_t* __restrict__ A, const bf16_t* __restrict__ W,
                const float* __restrict__ bias, float* __restrict__ C) {
  __shared__ bf16_t sW[64][D_];          // 128 KB panel, LDS offset 0
  (void)sW;
  const int lane = threadIdx.x & 31;
  const int wave = threadIdx.x >> 5;
  const int m0 = blockIdx.x * 128 + wave * 16;
  const int n0 = blockIdx.y * 64;

  if (threadIdx.x == 0) {
    tdm_load_panel(W + (size_t)n0 * D_, 0u, D_, NQKV_, D_, 64, D_);
    __builtin_amdgcn_s_wait_tensorcnt(0);
  }
  __syncthreads();
  const lds_bf16_t* sWp = lds_panel_ptr(0u);   // opaque view of TDM-written LDS

  v8f acc[4] = {};
  for (int k0 = 0; k0 < D_; k0 += 32) {
    const v16bf a = load_A_frag(A, D_, m0, k0, lane);
#pragma unroll
    for (int nt = 0; nt < 4; ++nt) {
      const v16bf bfrag = load_Bt_frag_lds(sWp, D_, k0, nt * 16, lane);
      acc[nt] = wmma_bf16(a, bfrag, acc[nt]);
    }
  }
  const int nlo = lane & 15;
  const int mhi = (lane >> 4) * 8;
#pragma unroll
  for (int nt = 0; nt < 4; ++nt) {
    const int n = n0 + nt * 16 + nlo;
    const float bv = bias[n];
#pragma unroll
    for (int r = 0; r < 8; ++r) {
      const int m = m0 + r + mhi;
      C[(size_t)m * NQKV_ + n] = acc[nt][r] + bv;
    }
  }
}

// ---------------- 4. RoPE + bf16 head layout -------------------------------

__global__ void __launch_bounds__(256)
rope_kernel(const float* __restrict__ qkv, const float* __restrict__ rope,
            bf16_t* __restrict__ qh, bf16_t* __restrict__ kh,
            bf16_t* __restrict__ vh) {
  const int id = blockIdx.x * blockDim.x + threadIdx.x;
  if (id >= BL_ * H_ * T_ * 32) return;
  const int dlo = id & 31;
  const int t   = (id >> 5) & 31;
  const int h   = (id >> 10) & 3;
  const int bl  = id >> 12;
  const size_t row = (size_t)bl * T_ + t;
  const float* base = qkv + row * NQKV_ + h * DH_;

  const float p = rope[t * DH_ + dlo];
  const float c = cosf(p);
  const float s = sinf(p);

  const float ql = base[dlo],            qu = base[dlo + 32];
  const float kl = base[INNER_ + dlo],   ku = base[INNER_ + dlo + 32];
  const float vl = base[2*INNER_ + dlo], vu = base[2*INNER_ + dlo + 32];

  const size_t ob = ((size_t)(bl * H_ + h)) * (T_ * DH_) + (size_t)t * DH_;
  qh[ob + dlo]      = (bf16_t)(ql * c - qu * s);
  qh[ob + dlo + 32] = (bf16_t)(qu * c + ql * s);
  kh[ob + dlo]      = (bf16_t)(kl * c - ku * s);
  kh[ob + dlo + 32] = (bf16_t)(ku * c + kl * s);
  vh[ob + dlo]      = (bf16_t)vl;
  vh[ob + dlo + 32] = (bf16_t)vu;
}

// ---------------- 5. Attention: one wave per (bl, head) --------------------

__global__ void __launch_bounds__(128)
attn_kernel(const bf16_t* __restrict__ qh, const bf16_t* __restrict__ kh,
            const bf16_t* __restrict__ vh, const int* __restrict__ mask,
            bf16_t* __restrict__ ao) {
  __shared__ float  sS[4][T_][33];
  __shared__ bf16_t sP[4][T_][34];
  const int lane = threadIdx.x & 31;
  const int wave = threadIdx.x >> 5;
  const int bl   = blockIdx.x;
  const int b    = bl >> 8;
  const size_t hb = ((size_t)(bl * H_ + wave)) * (T_ * DH_);
  const bf16_t* q = qh + hb;
  const bf16_t* k = kh + hb;
  const bf16_t* v = vh + hb;

  v8f s[2][2] = {};
  for (int kk = 0; kk < DH_; kk += 32) {
    const v16bf a0 = load_A_frag(q, DH_, 0,  kk, lane);
    const v16bf a1 = load_A_frag(q, DH_, 16, kk, lane);
    const v16bf b0 = load_Bt_frag(k, DH_, kk, 0,  lane);
    const v16bf b1 = load_Bt_frag(k, DH_, kk, 16, lane);
    s[0][0] = wmma_bf16(a0, b0, s[0][0]);
    s[0][1] = wmma_bf16(a0, b1, s[0][1]);
    s[1][0] = wmma_bf16(a1, b0, s[1][0]);
    s[1][1] = wmma_bf16(a1, b1, s[1][1]);
  }

  const int nlo = lane & 15;
  const int mhi = (lane >> 4) * 8;
#pragma unroll
  for (int mt = 0; mt < 2; ++mt)
#pragma unroll
    for (int nt = 0; nt < 2; ++nt)
#pragma unroll
      for (int r = 0; r < 8; ++r)
        sS[wave][mt * 16 + r + mhi][nt * 16 + nlo] = s[mt][nt][r];
  asm volatile("s_wait_dscnt 0" ::: "memory");

  const float scale = 0.125f;
  float rowv[T_];
  float mx = -__builtin_inff();
#pragma unroll
  for (int j = 0; j < T_; ++j) {
    const float biasv = mask[b * T_ + j] ? 0.0f : -__builtin_inff();
    const float tv = sS[wave][lane][j] * scale + biasv;
    rowv[j] = tv;
    mx = fmaxf(mx, tv);
  }
  float sum = 0.f;
#pragma unroll
  for (int j = 0; j < T_; ++j) {
    rowv[j] = __expf(rowv[j] - mx);
    sum += rowv[j];
  }
  const float invs = 1.0f / sum;
#pragma unroll
  for (int j = 0; j < T_; ++j)
    sP[wave][lane][j] = (bf16_t)(rowv[j] * invs);
  asm volatile("s_wait_dscnt 0" ::: "memory");

  const bf16_t* pmat = &sP[wave][0][0];
  const v16bf pa0 = load_A_frag(pmat, 34, 0,  0, lane);
  const v16bf pa1 = load_A_frag(pmat, 34, 16, 0, lane);
  v8f o[2][4] = {};
#pragma unroll
  for (int nt = 0; nt < 4; ++nt) {
    const v16bf bfrag = load_B_frag(v, DH_, 0, nt * 16, lane);
    o[0][nt] = wmma_bf16(pa0, bfrag, o[0][nt]);
    o[1][nt] = wmma_bf16(pa1, bfrag, o[1][nt]);
  }

#pragma unroll
  for (int mt = 0; mt < 2; ++mt)
#pragma unroll
    for (int nt = 0; nt < 4; ++nt)
#pragma unroll
      for (int r = 0; r < 8; ++r) {
        const int t = mt * 16 + r + mhi;
        const int d = wave * DH_ + nt * 16 + nlo;
        ao[((size_t)bl * T_ + t) * INNER_ + d] = (bf16_t)o[mt][nt][r];
      }
}

// ---------------- 6. Out proj GEMM (TDM-staged B panel) ---------------------

__global__ void __launch_bounds__(256)
out_gemm_kernel(const bf16_t* __restrict__ A, const bf16_t* __restrict__ W,
                float* __restrict__ out) {
  __shared__ bf16_t sW[64][INNER_];      // 32 KB panel, LDS offset 0
  (void)sW;
  const int lane = threadIdx.x & 31;
  const int wave = threadIdx.x >> 5;
  const int m0 = blockIdx.x * 128 + wave * 16;
  const int n0 = blockIdx.y * 64;

  if (threadIdx.x == 0) {
    tdm_load_panel(W + (size_t)n0 * INNER_, 0u, INNER_, D_, INNER_, 64, INNER_);
    __builtin_amdgcn_s_wait_tensorcnt(0);
  }
  __syncthreads();
  const lds_bf16_t* sWp = lds_panel_ptr(0u);

  v8f acc[4] = {};
  for (int k0 = 0; k0 < INNER_; k0 += 32) {
    const v16bf a = load_A_frag(A, INNER_, m0, k0, lane);
#pragma unroll
    for (int nt = 0; nt < 4; ++nt) {
      const v16bf bfrag = load_Bt_frag_lds(sWp, INNER_, k0, nt * 16, lane);
      acc[nt] = wmma_bf16(a, bfrag, acc[nt]);
    }
  }
  const int nlo = lane & 15;
  const int mhi = (lane >> 4) * 8;
#pragma unroll
  for (int nt = 0; nt < 4; ++nt) {
    const int n = n0 + nt * 16 + nlo;
#pragma unroll
    for (int r = 0; r < 8; ++r) {
      const int m = m0 + r + mhi;
      const int bl = m >> 5, t = m & 31;
      const int b = bl >> 8, l = bl & 255;
      out[(((size_t)(b * T_ + t)) * L_ + l) * D_ + n] = acc[nt][r];
    }
  }
}

// ---------------- Host launcher --------------------------------------------

extern "C" void kernel_launch(void* const* d_in, const int* in_sizes, int n_in,
                              void* d_out, int out_size, void* d_ws, size_t ws_size,
                              hipStream_t stream) {
  const float* x      = (const float*)d_in[0];
  const float* rope   = (const float*)d_in[1];
  const int*   mask   = (const int*)d_in[2];
  const float* gamma  = (const float*)d_in[3];
  const float* beta   = (const float*)d_in[4];
  const float* w_qkv  = (const float*)d_in[5];
  const float* b_qkv  = (const float*)d_in[6];
  const float* w_out  = (const float*)d_in[7];
  float* out = (float*)d_out;
  char*  ws  = (char*)d_ws;

  const size_t WQKV_OFF = 0;
  const size_t WOUT_OFF = (size_t)NQKV_ * D_ * 2;
  const size_t XN_OFF   = WOUT_OFF + (size_t)D_ * INNER_ * 2;
  const size_t XN_BYTES = (size_t)ROWS_ * D_ * 2;
  const size_t QKV_OFF  = XN_OFF + XN_BYTES;

  bf16_t* wqkvB = (bf16_t*)(ws + WQKV_OFF);
  bf16_t* woutB = (bf16_t*)(ws + WOUT_OFF);
  bf16_t* xn    = (bf16_t*)(ws + XN_OFF);
  bf16_t* qh    = xn;
  bf16_t* kh    = qh + (size_t)ROWS_ * INNER_;
  bf16_t* vh    = kh + (size_t)ROWS_ * INNER_;
  bf16_t* ao    = vh + (size_t)ROWS_ * INNER_;
  float*  qkv   = (float*)(ws + QKV_OFF);

  convert_w_kernel<<<(NQKV_ * D_ + 255) / 256, 256, 0, stream>>>(w_qkv, w_out,
                                                                 wqkvB, woutB);
  ln_kernel<<<ROWS_, 256, 0, stream>>>(x, gamma, beta, xn);
  qkv_gemm_kernel<<<dim3(ROWS_ / 128, NQKV_ / 64), 256, 0, stream>>>(xn, wqkvB,
                                                                     b_qkv, qkv);
  rope_kernel<<<(BL_ * H_ * T_ * 32) / 256, 256, 0, stream>>>(qkv, rope,
                                                              qh, kh, vh);
  attn_kernel<<<BL_, 128, 0, stream>>>(qh, kh, vh, mask, ao);
  out_gemm_kernel<<<dim3(ROWS_ / 128, D_ / 64), 256, 0, stream>>>(ao, woutB, out);
}
